// EPMixtralMoE_48722109006442
// MI455X (gfx1250) — compile-verified, matrix operand order checked by reading, not run on hardware
//
#include <hip/hip_runtime.h>
#include <hip/hip_bf16.h>
#include <math.h>

typedef __bf16 bf16_t;
typedef __attribute__((ext_vector_type(16))) __bf16 v16bf;
typedef __attribute__((ext_vector_type(8)))  __bf16 v8bf;
typedef __attribute__((ext_vector_type(4)))  __bf16 v4bf;
typedef __attribute__((ext_vector_type(8)))  float  v8f;
typedef __attribute__((ext_vector_type(4)))  int    v4i;

#define S_TOK 4096
#define H_DIM 2048
#define I_DIM 7168
#define NEXP  8
#define CAP_  2048
#define APAD  40   // LDS row stride (32 data + 8 pad bf16) -> 80B rows, bank-spread

#if defined(__gfx1250__) && __has_builtin(__builtin_amdgcn_global_load_async_to_lds_b128)
#define HAVE_ASYNC_LDS 1
#else
#define HAVE_ASYNC_LDS 0
#endif

#if HAVE_ASYNC_LDS
#define ASYNC_WAIT() asm volatile("s_wait_asynccnt 0" ::: "memory")
typedef __attribute__((address_space(1))) v4i gv4i;   // global v4i
typedef __attribute__((address_space(3))) v4i lv4i;   // LDS v4i
#endif

// ---------------- helpers ----------------

__device__ __forceinline__ v4bf cvt4(float4 f) {
  v4bf r;
  r[0] = (bf16_t)f.x; r[1] = (bf16_t)f.y; r[2] = (bf16_t)f.z; r[3] = (bf16_t)f.w;
  return r;
}

// gfx1250 16-bit A/B fragment layout: lanes 0-15 hold K={0..7,16..23},
// lanes 16-31 hold K={8..15,24..31}.  Two 16B LDS reads per fragment.
__device__ __forceinline__ v16bf ldsFrag(const bf16_t* base, int row, int kh) {
  const v8bf lo = *(const v8bf*)(base + row * APAD + kh);
  const v8bf hi = *(const v8bf*)(base + row * APAD + kh + 16);
  v16bf f;
#pragma unroll
  for (int i = 0; i < 8; i++) { f[i] = lo[i]; f[i + 8] = hi[i]; }
  return f;
}

// ---------------- 1. gating: logits, top-2, renormalized weights ----------------

__global__ __launch_bounds__(256) void moe_gate(const float* __restrict__ x,
                                                const float* __restrict__ gw,
                                                int* __restrict__ ids,
                                                float* __restrict__ wts) {
  const int wave = threadIdx.x >> 5, lane = threadIdx.x & 31;
  const int s = blockIdx.x * 8 + wave;
  float acc[NEXP];
#pragma unroll
  for (int e = 0; e < NEXP; e++) acc[e] = 0.f;
  const float* xr = x + (size_t)s * H_DIM;
  for (int h = lane; h < H_DIM; h += 32) {
    const float xv = xr[h];
#pragma unroll
    for (int e = 0; e < NEXP; e++) acc[e] += xv * gw[e * H_DIM + h];
  }
#pragma unroll
  for (int e = 0; e < NEXP; e++) {
#pragma unroll
    for (int off = 16; off > 0; off >>= 1) acc[e] += __shfl_xor(acc[e], off, 32);
  }
  if (lane == 0) {
    int b1 = 0; float l1 = acc[0];
#pragma unroll
    for (int e = 1; e < NEXP; e++) if (acc[e] > l1) { l1 = acc[e]; b1 = e; }
    int b2 = (b1 == 0) ? 1 : 0; float l2 = acc[b2];
#pragma unroll
    for (int e = 0; e < NEXP; e++)
      if (e != b1 && e != b2 && acc[e] > l2) { l2 = acc[e]; b2 = e; }
    const float w1 = 1.f / (1.f + __expf(l2 - l1));   // top-2 renorm == 2-way softmax
    ids[2 * s] = b1; ids[2 * s + 1] = b2;
    wts[2 * s] = w1; wts[2 * s + 1] = 1.f - w1;
  }
}

// ---------------- 2. routing: exact cumsum slot assignment ----------------

__global__ __launch_bounds__(256) void moe_route(const int* __restrict__ ids,
                                                 int* __restrict__ slots,
                                                 int* __restrict__ n_used) {
  __shared__ int cnt1[256][NEXP];
  __shared__ int cnt2[256][NEXP];
  __shared__ int tot1[NEXP], tot2[NEXP];
  const int t = threadIdx.x;
  int c1[NEXP], c2[NEXP];
#pragma unroll
  for (int e = 0; e < NEXP; e++) { c1[e] = 0; c2[e] = 0; }
  const int base = t * 16;
  for (int i = 0; i < 16; i++) {
    const int s = base + i;
    c1[ids[2 * s]]++;
    c2[ids[2 * s + 1]]++;
  }
#pragma unroll
  for (int e = 0; e < NEXP; e++) { cnt1[t][e] = c1[e]; cnt2[t][e] = c2[e]; }
  __syncthreads();
  if (t < NEXP) {
    int run = 0;
    for (int i = 0; i < 256; i++) { const int v = cnt1[i][t]; cnt1[i][t] = run; run += v; }
    tot1[t] = run;
  } else if (t < 2 * NEXP) {
    const int e = t - NEXP; int run = 0;
    for (int i = 0; i < 256; i++) { const int v = cnt2[i][e]; cnt2[i][e] = run; run += v; }
    tot2[e] = run;
  }
  __syncthreads();
  int off1[NEXP], off2[NEXP];
#pragma unroll
  for (int e = 0; e < NEXP; e++) { off1[e] = cnt1[t][e]; off2[e] = cnt2[t][e]; }
  for (int i = 0; i < 16; i++) {
    const int s = base + i;
    const int e1 = ids[2 * s], e2 = ids[2 * s + 1];
    slots[2 * s]     = off1[e1]++;
    slots[2 * s + 1] = tot1[e2] + off2[e2]++;   // 2nd choices after ALL 1st choices
  }
  if (t < NEXP) {
    const int u = tot1[t] + tot2[t];
    n_used[t] = u < CAP_ ? u : CAP_;
  }
}

// ---------------- 3. scatter tokens into (E, CAP, H) bf16 ----------------

__global__ __launch_bounds__(256) void moe_scatter(const float* __restrict__ x,
                                                   const int* __restrict__ ids,
                                                   const int* __restrict__ slots,
                                                   bf16_t* __restrict__ disp) {
  const int s = blockIdx.x;
  const int e1 = ids[2 * s], e2 = ids[2 * s + 1];
  const int s1 = slots[2 * s], s2 = slots[2 * s + 1];
  const float* xr = x + (size_t)s * H_DIM;
  const int j = threadIdx.x * 8;
  const float4 f0 = *(const float4*)(xr + j);
  const float4 f1 = *(const float4*)(xr + j + 4);
  const v4bf b0 = cvt4(f0), b1v = cvt4(f1);
  if (s1 < CAP_) {
    bf16_t* d = disp + ((size_t)e1 * CAP_ + s1) * H_DIM + j;
    *(v4bf*)(d) = b0; *(v4bf*)(d + 4) = b1v;
  }
  if (s2 < CAP_) {
    bf16_t* d = disp + ((size_t)e2 * CAP_ + s2) * H_DIM + j;
    *(v4bf*)(d) = b0; *(v4bf*)(d + 4) = b1v;
  }
}

// ---------------- 4. GEMM1: act = silu(X W1^T) * (X W3^T) ----------------
// Block tile 128M x 64N, 8 waves as 2(M) x 4(N); per-wave 64x16 dual output:
// 8 accumulator tiles = 64 VGPRs (no spills).  A tile staged via async
// global->LDS copy (ASYNCcnt); B tiles converted fp32->bf16 through VGPRs.

__global__ __launch_bounds__(256) void moe_gemm1(const bf16_t* __restrict__ disp,
                                                 const float* __restrict__ w13,
                                                 const int* __restrict__ n_used,
                                                 bf16_t* __restrict__ act) {
  const int e  = blockIdx.z;
  const int m0 = blockIdx.y * 128;
  const int n0 = blockIdx.x * 64;
  if (m0 >= n_used[e]) return;

  __shared__ __align__(16) bf16_t sA [2][128 * APAD];
  __shared__ __align__(16) bf16_t sB1[2][64 * APAD];
  __shared__ __align__(16) bf16_t sB3[2][64 * APAD];

  const int tid  = threadIdx.x;
  const int lane = tid & 31;
  const int wave = tid >> 5;
  const int wm = (wave & 1) * 64;
  const int wn = (wave >> 1) * 16;
  const int rl = lane & 15;
  const int kh = (lane >> 4) * 8;

  const bf16_t* Ag  = disp + ((size_t)e * CAP_ + m0) * H_DIM;
  const float*  B1g = w13  + ((size_t)e * 2 * I_DIM + n0) * H_DIM;
  const float*  B3g = B1g + (size_t)I_DIM * H_DIM;

#if !HAVE_ASYNC_LDS
  uint4 aR[2];
#endif
  float4 b1R[2], b3R[2];

  auto issueA = [&](int k0, int buf) {
#pragma unroll
    for (int i = 0; i < 2; i++) {
      const int c = tid + 256 * i;
      const int r = c >> 2, kc = (c & 3) * 8;
#if HAVE_ASYNC_LDS
      __builtin_amdgcn_global_load_async_to_lds_b128(
          (gv4i*)(Ag + (size_t)r * H_DIM + k0 + kc),
          (lv4i*)(&sA[buf][r * APAD + kc]), 0, 0);
#else
      aR[i] = *(const uint4*)(Ag + (size_t)r * H_DIM + k0 + kc);
#endif
    }
  };
  auto loadB = [&](int k0) {
#pragma unroll
    for (int i = 0; i < 2; i++) {
      const int c = tid + 256 * i;
      const size_t o = (size_t)(c >> 3) * H_DIM + k0 + (c & 7) * 4;
      b1R[i] = *(const float4*)(B1g + o);
      b3R[i] = *(const float4*)(B3g + o);
    }
  };
  auto storeL = [&](int buf) {
#if !HAVE_ASYNC_LDS
#pragma unroll
    for (int i = 0; i < 2; i++) {
      const int c = tid + 256 * i;
      *(uint4*)(&sA[buf][(c >> 2) * APAD + (c & 3) * 8]) = aR[i];
    }
#endif
#pragma unroll
    for (int i = 0; i < 2; i++) {
      const int c = tid + 256 * i;
      const int o = (c >> 3) * APAD + (c & 7) * 4;
      *(v4bf*)(&sB1[buf][o]) = cvt4(b1R[i]);
      *(v4bf*)(&sB3[buf][o]) = cvt4(b3R[i]);
    }
  };

  v8f cg[4], cu[4];
#pragma unroll
  for (int i = 0; i < 4; i++) { cg[i] = v8f{}; cu[i] = v8f{}; }

  issueA(0, 0);
  loadB(0);
  storeL(0);
#if HAVE_ASYNC_LDS
  ASYNC_WAIT();
#endif
  const int NK = H_DIM / 32;
  for (int kt = 0; kt < NK; ++kt) {
    __syncthreads();
    const int buf = kt & 1;
    if (kt + 1 < NK) {
      issueA((kt + 1) * 32, buf ^ 1);   // async copy overlaps WMMA below
      loadB((kt + 1) * 32);
    }
    v16bf af[4], b1f, b3f;
#pragma unroll
    for (int i = 0; i < 4; i++) af[i] = ldsFrag(sA[buf], wm + i * 16 + rl, kh);
    b1f = ldsFrag(sB1[buf], wn + rl, kh);
    b3f = ldsFrag(sB3[buf], wn + rl, kh);
#pragma unroll
    for (int i = 0; i < 4; i++) {
      cg[i] = __builtin_amdgcn_wmma_f32_16x16x32_bf16(false, af[i], false, b1f,
                                                      (short)0, cg[i], false, false);
      cu[i] = __builtin_amdgcn_wmma_f32_16x16x32_bf16(false, af[i], false, b3f,
                                                      (short)0, cu[i], false, false);
    }
    if (kt + 1 < NK) {
      storeL(buf ^ 1);
#if HAVE_ASYNC_LDS
      ASYNC_WAIT();                     // own async A writes done before barrier
#endif
    }
  }

  bf16_t* actE = act + (size_t)e * CAP_ * I_DIM;
  const int mh = (lane >> 4) * 8;
  const int nn = n0 + wn + (lane & 15);
#pragma unroll
  for (int i = 0; i < 4; i++) {
#pragma unroll
    for (int r = 0; r < 8; r++) {
      const int mm = m0 + wm + i * 16 + mh + r;
      const float g = cg[i][r];
      const float u = cu[i][r];
      const float a = (g / (1.f + __expf(-g))) * u;
      actE[(size_t)mm * I_DIM + nn] = (bf16_t)a;
    }
  }
}

// ---------------- 5. GEMM2: y = act W2^T ----------------
// Block tile 128x128, per-wave 64x32, 8 accumulator tiles; async A staging.

__global__ __launch_bounds__(256) void moe_gemm2(const bf16_t* __restrict__ actA,
                                                 const float* __restrict__ w2,
                                                 const int* __restrict__ n_used,
                                                 float* __restrict__ y) {
  const int e  = blockIdx.z;
  const int m0 = blockIdx.y * 128;
  const int n0 = blockIdx.x * 128;
  if (m0 >= n_used[e]) return;

  __shared__ __align__(16) bf16_t sA[2][128 * APAD];
  __shared__ __align__(16) bf16_t sB[2][128 * APAD];

  const int tid  = threadIdx.x;
  const int lane = tid & 31;
  const int wave = tid >> 5;
  const int wm = (wave & 1) * 64;
  const int wn = (wave >> 1) * 32;
  const int rl = lane & 15;
  const int kh = (lane >> 4) * 8;

  const bf16_t* Ag = actA + ((size_t)e * CAP_ + m0) * I_DIM;
  const float*  Bg = w2   + ((size_t)e * H_DIM + n0) * I_DIM;

#if !HAVE_ASYNC_LDS
  uint4 aR[2];
#endif
  float4 bR[4];

  auto issueA = [&](int k0, int buf) {
#pragma unroll
    for (int i = 0; i < 2; i++) {
      const int c = tid + 256 * i;
      const int r = c >> 2, kc = (c & 3) * 8;
#if HAVE_ASYNC_LDS
      __builtin_amdgcn_global_load_async_to_lds_b128(
          (gv4i*)(Ag + (size_t)r * I_DIM + k0 + kc),
          (lv4i*)(&sA[buf][r * APAD + kc]), 0, 0);
#else
      aR[i] = *(const uint4*)(Ag + (size_t)r * I_DIM + k0 + kc);
#endif
    }
  };
  auto loadB = [&](int k0) {
#pragma unroll
    for (int i = 0; i < 4; i++) {
      const int c = tid + 256 * i;
      bR[i] = *(const float4*)(Bg + (size_t)(c >> 3) * I_DIM + k0 + (c & 7) * 4);
    }
  };
  auto storeL = [&](int buf) {
#if !HAVE_ASYNC_LDS
#pragma unroll
    for (int i = 0; i < 2; i++) {
      const int c = tid + 256 * i;
      *(uint4*)(&sA[buf][(c >> 2) * APAD + (c & 3) * 8]) = aR[i];
    }
#endif
#pragma unroll
    for (int i = 0; i < 4; i++) {
      const int c = tid + 256 * i;
      *(v4bf*)(&sB[buf][(c >> 3) * APAD + (c & 7) * 4]) = cvt4(bR[i]);
    }
  };

  v8f cc[4][2];
#pragma unroll
  for (int i = 0; i < 4; i++)
#pragma unroll
    for (int j = 0; j < 2; j++) cc[i][j] = v8f{};

  issueA(0, 0);
  loadB(0);
  storeL(0);
#if HAVE_ASYNC_LDS
  ASYNC_WAIT();
#endif
  const int NK = I_DIM / 32;
  for (int kt = 0; kt < NK; ++kt) {
    __syncthreads();
    const int buf = kt & 1;
    if (kt + 1 < NK) {
      issueA((kt + 1) * 32, buf ^ 1);
      loadB((kt + 1) * 32);
    }
    v16bf af[4], bf[2];
#pragma unroll
    for (int i = 0; i < 4; i++) af[i] = ldsFrag(sA[buf], wm + i * 16 + rl, kh);
#pragma unroll
    for (int j = 0; j < 2; j++) bf[j] = ldsFrag(sB[buf], wn + j * 16 + rl, kh);
#pragma unroll
    for (int i = 0; i < 4; i++)
#pragma unroll
      for (int j = 0; j < 2; j++)
        cc[i][j] = __builtin_amdgcn_wmma_f32_16x16x32_bf16(false, af[i], false, bf[j],
                                                           (short)0, cc[i][j], false, false);
    if (kt + 1 < NK) {
      storeL(buf ^ 1);
#if HAVE_ASYNC_LDS
      ASYNC_WAIT();
#endif
    }
  }

  float* yE = y + (size_t)e * CAP_ * H_DIM;
  const int mh = (lane >> 4) * 8;
  const int nc = lane & 15;
#pragma unroll
  for (int i = 0; i < 4; i++)
#pragma unroll
    for (int j = 0; j < 2; j++) {
      const int nn = n0 + wn + j * 16 + nc;
#pragma unroll
      for (int r = 0; r < 8; r++) {
        const int mm = m0 + wm + i * 16 + mh + r;
        yE[(size_t)mm * H_DIM + nn] = cc[i][j][r];
      }
    }
}

// ---------------- 6. combine: weighted gather with capacity-drop guards ----------------

__global__ __launch_bounds__(256) void moe_combine(const float* __restrict__ y,
                                                   const int* __restrict__ ids,
                                                   const int* __restrict__ slots,
                                                   const float* __restrict__ wts,
                                                   float* __restrict__ out) {
  const int s = blockIdx.x;
  const int e1 = ids[2 * s], e2 = ids[2 * s + 1];
  const int s1 = slots[2 * s], s2 = slots[2 * s + 1];
  const float w1 = wts[2 * s], w2 = wts[2 * s + 1];
  const bool v1 = s1 < CAP_, v2 = s2 < CAP_;
  const float* y1 = y + ((size_t)e1 * CAP_ + (v1 ? s1 : 0)) * H_DIM;
  const float* y2 = y + ((size_t)e2 * CAP_ + (v2 ? s2 : 0)) * H_DIM;
  const int j = threadIdx.x * 8;
  float acc[8];
#pragma unroll
  for (int k = 0; k < 8; k++) acc[k] = 0.f;
  if (v1) {
    const float4 a = *(const float4*)(y1 + j);
    const float4 b = *(const float4*)(y1 + j + 4);
    acc[0] += w1 * a.x; acc[1] += w1 * a.y; acc[2] += w1 * a.z; acc[3] += w1 * a.w;
    acc[4] += w1 * b.x; acc[5] += w1 * b.y; acc[6] += w1 * b.z; acc[7] += w1 * b.w;
  }
  if (v2) {
    const float4 a = *(const float4*)(y2 + j);
    const float4 b = *(const float4*)(y2 + j + 4);
    acc[0] += w2 * a.x; acc[1] += w2 * a.y; acc[2] += w2 * a.z; acc[3] += w2 * a.w;
    acc[4] += w2 * b.x; acc[5] += w2 * b.y; acc[6] += w2 * b.z; acc[7] += w2 * b.w;
  }
  float* o = out + (size_t)s * H_DIM + j;
#pragma unroll
  for (int k = 0; k < 8; k++) o[k] = acc[k];
}

// ---------------- launch ----------------

extern "C" void kernel_launch(void* const* d_in, const int* in_sizes, int n_in,
                              void* d_out, int out_size, void* d_ws, size_t ws_size,
                              hipStream_t stream) {
  (void)in_sizes; (void)n_in; (void)out_size; (void)ws_size;
  const float* x   = (const float*)d_in[0];
  const float* gw  = (const float*)d_in[1];
  const float* w13 = (const float*)d_in[2];
  const float* w2  = (const float*)d_in[3];
  float* out = (float*)d_out;

  char* ws = (char*)d_ws;
  size_t off = 0;
  auto alloc = [&](size_t bytes) -> void* {
    void* p = ws + off;
    off += bytes;
    off = (off + 255) & ~(size_t)255;
    return p;
  };

  int*    ids    = (int*)alloc((size_t)S_TOK * 2 * sizeof(int));
  int*    slots  = (int*)alloc((size_t)S_TOK * 2 * sizeof(int));
  float*  wts    = (float*)alloc((size_t)S_TOK * 2 * sizeof(float));
  int*    n_used = (int*)alloc((size_t)NEXP * sizeof(int));
  bf16_t* disp   = (bf16_t*)alloc((size_t)NEXP * CAP_ * H_DIM * sizeof(bf16_t));
  bf16_t* act    = (bf16_t*)alloc((size_t)NEXP * CAP_ * I_DIM * sizeof(bf16_t));
  float*  yb     = (float*)alloc((size_t)NEXP * CAP_ * H_DIM * sizeof(float));

  moe_gate   <<<S_TOK / 8, 256, 0, stream>>>(x, gw, ids, wts);
  moe_route  <<<1,         256, 0, stream>>>(ids, slots, n_used);
  moe_scatter<<<S_TOK,     256, 0, stream>>>(x, ids, slots, disp);
  moe_gemm1  <<<dim3(I_DIM / 64,  CAP_ / 128, NEXP), 256, 0, stream>>>(disp, w13, n_used, act);
  moe_gemm2  <<<dim3(H_DIM / 128, CAP_ / 128, NEXP), 256, 0, stream>>>(act, w2, n_used, yb);
  moe_combine<<<S_TOK,     256, 0, stream>>>(yb, ids, slots, wts, out);
}
